// MoELayer_56599079027008
// MI455X (gfx1250) — matrix operand order repaired and checked by reading, not compiled, not run on hardware
//
#include <hip/hip_runtime.h>
#include <hip/hip_bf16.h>

#define D_MODEL 1024
#define HIDDEN  4096
#define NEXP    8
#define NTOK    8192
#define CAP     8192   // worst case: every token routed to one expert

typedef __attribute__((ext_vector_type(16))) __bf16        v16bf;
typedef __attribute__((ext_vector_type(8)))  float         v8f;
typedef __attribute__((ext_vector_type(4)))  unsigned int  v4u;
typedef __attribute__((ext_vector_type(8)))  int           v8i;
typedef __attribute__((ext_vector_type(4)))  int           v4i;

union ABf {
    v16bf v;
    __bf16 h[16];
    unsigned int u[8];
};

// Pack A fragment (16x32 bf16, row-major LDS source with given stride) per the
// CDNA5 16-bit A layout:
//   lanes 0-15 : M=lane,    K = k0+{0..7} in v0..3, k0+{16..23} in v4..7
//   lanes 16-31: M=lane-16, K = k0+{8..15}          / k0+{24..31}
// Lowers to 2x ds_load_b128.
__device__ __forceinline__ v16bf packA(const __bf16* base, int stride, int k0, int lane) {
    ABf r;
    const int m  = lane & 15;
    const int kb = (lane & 16) ? 8 : 0;
    const __bf16* row = base + m * stride;
#pragma unroll
    for (int j = 0; j < 4; ++j) {
        r.u[j]     = *(const unsigned int*)(row + k0 + kb + 2 * j);
        r.u[j + 4] = *(const unsigned int*)(row + k0 + 16 + kb + 2 * j);
    }
    return r.v;
}

// Load a pre-swizzled B fragment: fragment = 32 lanes x 32 contiguous bytes
// (1 KB per 32x16 K-x-N tile). Two b128 loads per lane, fully coalesced.
__device__ __forceinline__ v16bf loadB(const unsigned int* __restrict__ Ws,
                                       long frag, int lane) {
    union { v16bf v; uint4 q[2]; } r;
    const uint4* p = (const uint4*)(Ws + frag * 256 + lane * 8);
    r.q[0] = p[0];
    r.q[1] = p[1];
    return r.v;
}

// ---------------------------------------------------------------- zero -----
__global__ void moe_zero_kernel(float* __restrict__ out, int out_n,
                                unsigned int* __restrict__ wshdr) {
    const int i = blockIdx.x * blockDim.x + threadIdx.x;
    if (i < out_n) out[i] = 0.0f;
    if (i < 64)   wshdr[i] = 0u;   // counts / probs_sum / assign_sum region
}

// ----------------------------------------------------- weight swizzle ------
// fp32 W[e][k][n] (leading dim ldw, K-extent kdim)  ->  bf16 fragment-major
// layout matching loadB: lanes 0-15 hold K=0..7,16..23 of a 32x16 tile, lanes 16-31
// hold K=8..15,24..31; dword dw is the K-pair for column n0+(lane&15).
__global__ void moe_swz_kernel(const float* __restrict__ W,
                               unsigned int* __restrict__ Ws,
                               int kdim, int ldw, long total) {
    const long stride = (long)gridDim.x * blockDim.x;
    const int  kblks  = kdim >> 5;
    const int  nblks  = ldw >> 4;
    for (long t = (long)blockIdx.x * blockDim.x + threadIdx.x; t < total; t += stride) {
        const int  dw   = (int)(t & 7);
        const int  lane = (int)((t >> 3) & 31);
        const long frag = t >> 8;
        const int  kblk = (int)(frag % kblks);
        const long rem  = frag / kblks;
        const int  nblk = (int)(rem % nblks);
        const int  e    = (int)(rem / nblks);
        const int  n    = nblk * 16 + (lane & 15);
        const int  k    = kblk * 32 + ((dw & 4) ? 16 : 0) + ((lane & 16) ? 8 : 0) + 2 * (dw & 3);
        const float* src = W + (size_t)e * kdim * ldw + (size_t)k * ldw + n;
        union { unsigned int u; __bf16 h[2]; } r;
        r.h[0] = (__bf16)src[0];
        r.h[1] = (__bf16)src[ldw];
        Ws[t] = r.u;
    }
}

// ------------------------------------------------------- x -> bf16 ---------
__global__ void moe_cvtx_kernel(const float* __restrict__ x,
                                unsigned int* __restrict__ xs, long total_dw) {
    const long stride = (long)gridDim.x * blockDim.x;
    for (long i = (long)blockIdx.x * blockDim.x + threadIdx.x; i < total_dw; i += stride) {
        union { unsigned int u; __bf16 h[2]; } r;
        r.h[0] = (__bf16)x[2 * i];
        r.h[1] = (__bf16)x[2 * i + 1];
        xs[i] = r.u;
    }
}

// ---------------------------------------------------------------- gate -----
__global__ void moe_gate_kernel(const float* __restrict__ x,
                                const float* __restrict__ Wg,
                                const float* __restrict__ bg,
                                int*   __restrict__ counts,
                                float* __restrict__ probs_sum,
                                float* __restrict__ assign_sum,
                                int*   __restrict__ perm,
                                float* __restrict__ gatebuf) {
    const int n = blockIdx.x * blockDim.x + threadIdx.x;
    if (n >= NTOK) return;

    float acc[NEXP];
#pragma unroll
    for (int e = 0; e < NEXP; ++e) acc[e] = bg[e];

    const float* xr = x + (size_t)n * D_MODEL;
    for (int k = 0; k < D_MODEL; ++k) {
        const float xv = xr[k];
#pragma unroll
        for (int e = 0; e < NEXP; ++e) acc[e] += xv * Wg[k * NEXP + e];
    }

    // full softmax (for expected_prob metadata)
    float mx = acc[0];
#pragma unroll
    for (int e = 1; e < NEXP; ++e) mx = fmaxf(mx, acc[e]);
    float p[NEXP], s = 0.0f;
#pragma unroll
    for (int e = 0; e < NEXP; ++e) { p[e] = __expf(acc[e] - mx); s += p[e]; }
    const float inv = 1.0f / s;
#pragma unroll
    for (int e = 0; e < NEXP; ++e) atomicAdd(&probs_sum[e], p[e] * inv);

    // top-2 (first occurrence wins ties, matching lax.top_k)
    int e0 = 0;
#pragma unroll
    for (int e = 1; e < NEXP; ++e) if (acc[e] > acc[e0]) e0 = e;
    int e1 = (e0 == 0) ? 1 : 0;
#pragma unroll
    for (int e = 0; e < NEXP; ++e)
        if (e != e0 && acc[e] > acc[e1]) e1 = e;

    // softmax over the two selected logits (l0 >= l1)
    const float t  = __expf(acc[e1] - acc[e0]);
    const float g0 = 1.0f / (1.0f + t);
    const float g1 = t / (1.0f + t);

    const int s0 = atomicAdd(&counts[e0], 1);
    perm[e0 * CAP + s0]    = n;
    gatebuf[e0 * CAP + s0] = g0;
    atomicAdd(&assign_sum[e0], g0);

    const int s1 = atomicAdd(&counts[e1], 1);
    perm[e1 * CAP + s1]    = n;
    gatebuf[e1 * CAP + s1] = g1;
    atomicAdd(&assign_sum[e1], g1);
}

// ------------------------------------------------------------ finalize -----
__global__ void moe_finalize_kernel(const int* __restrict__ counts,
                                    const float* __restrict__ probs_sum,
                                    const float* __restrict__ assign_sum,
                                    float* __restrict__ meta) {
    const int t = threadIdx.x;
    if (t < NEXP) {
        meta[1 + t]            = (float)counts[t];             // tokens_per_expert
        meta[1 + NEXP + t]     = probs_sum[t] / (float)NTOK;   // expected_prob
        meta[1 + 2 * NEXP + t] = assign_sum[t] / (float)NTOK;  // assignment_fraction
    }
    if (t == 0) {
        float loss = 0.0f;
        for (int e = 0; e < NEXP; ++e)
            loss += (probs_sum[e] / (float)NTOK) * (assign_sum[e] / (float)NTOK);
        meta[0] = loss * (float)NEXP;
    }
}

// --------------------------------------------------------------- FFN -------
// grid = (CAP/16, NEXP); block = 256 (8 waves). One WG = one 16-token tile of
// one expert. The gathered X tile is pulled into LDS by the Tensor Data Mover
// in gather mode (16 row indices from the routing perm, bf16 token tensor),
// then fused FFN1 -> ReLU -> FFN2 over 128-wide hidden chunks with the
// 16x1024 output tile resident in WMMA accumulators (64 VGPRs/lane).
// Hlds is double-buffered: one workgroup barrier per hidden chunk.
__global__ __launch_bounds__(256, 1)
void moe_ffn_kernel(const unsigned int* __restrict__ xs,   // bf16 tokens (NTOK x D)
                    const int*   __restrict__ perm,
                    const float* __restrict__ gatebuf,
                    const int*   __restrict__ counts,
                    const unsigned int* __restrict__ W1s,
                    const float* __restrict__ b1,
                    const unsigned int* __restrict__ W2s,
                    const float* __restrict__ b2,
                    float* __restrict__ out) {
    const int e    = blockIdx.y;
    const int cnt  = counts[e];
    const int row0 = blockIdx.x * 16;
    if (row0 >= cnt) return;

    __shared__ __align__(16) __bf16 Xlds[16 * D_MODEL];   // 32 KB
    __shared__ __align__(16) __bf16 Hlds[2][16 * 128];    // 8 KB, double-buffered
    __shared__ int   trow[16];
    __shared__ float grow[16];

    const int tid  = threadIdx.x;
    const int wave = tid >> 5;
    const int lane = tid & 31;

    if (tid < 16) {
        const int r = row0 + tid;
        if (r < cnt) {
            trow[tid] = perm[e * CAP + r];
            grow[tid] = gatebuf[e * CAP + r];
        } else {
            trow[tid] = 0;     // pad rows gather token 0; gate 0 kills them
            grow[tid] = 0.0f;
        }
    }
    __syncthreads();

    // ---- TDM gather: 16 rows x 1024 bf16 from xs into Xlds -----------------
    if (wave == 0) {
        const unsigned long long ga = (unsigned long long)(uintptr_t)xs;
        v4u g0;
        g0.x = 1u | (1u << 31);                          // count=1, gather_mode, 16-bit idx
        g0.y = (unsigned)(uintptr_t)(&Xlds[0]);          // lds_addr
        g0.z = (unsigned)(ga & 0xffffffffu);             // global_addr[31:0]
        g0.w = (unsigned)((ga >> 32) & 0x01ffffffu) | (2u << 30);  // addr[56:32], type=2

        v8i g1;
        g1[0] = (int)(1u << 16);        // data_size = 2 bytes
        g1[1] = (int)(1024u << 16);     // tensor_dim0 = 1024 (bits 63:48)
        g1[2] = (int)(8192u << 16);     // tensor_dim1 = NTOK (bits 95:80)
        g1[3] = (int)(1024u << 16);     // tile_dim0 = 1024 (bits 127:112)
        g1[4] = 16;                     // tile_dim1 = #valid indices
        g1[5] = 1024;                   // tensor_dim0_stride
        g1[6] = 0;
        g1[7] = 0;

        unsigned pk[8];
#pragma unroll
        for (int r = 0; r < 8; ++r)
            pk[r] = ((unsigned)(unsigned short)trow[2 * r]) |
                    (((unsigned)(unsigned short)trow[2 * r + 1]) << 16);
        v4i g2; g2[0] = (int)pk[0]; g2[1] = (int)pk[1]; g2[2] = (int)pk[2]; g2[3] = (int)pk[3];
        v4i g3; g3[0] = (int)pk[4]; g3[1] = (int)pk[5]; g3[2] = (int)pk[6]; g3[3] = (int)pk[7];

#if defined(__clang_major__) && __clang_major__ >= 23
        __builtin_amdgcn_tensor_load_to_lds(g0, g1, g2, g3,
                                            (v8i){0, 0, 0, 0, 0, 0, 0, 0}, 0);
#else
        __builtin_amdgcn_tensor_load_to_lds(g0, g1, g2, g3, 0);
#endif
        __builtin_amdgcn_s_wait_tensorcnt((short)0);
    }
    __syncthreads();

    v8f cacc[8];
#pragma unroll
    for (int s = 0; s < 8; ++s) cacc[s] = (v8f){0.f,0.f,0.f,0.f,0.f,0.f,0.f,0.f};

    for (int hc = 0; hc < HIDDEN / 128; ++hc) {
        __bf16* Hbuf = Hlds[hc & 1];

        // ---- FFN1: this wave computes h[:, hc*128 + wave*16 .. +16), K = 1024
        const int hcol0 = hc * 128 + wave * 16;
        const unsigned int* W1f =
            W1s + (((long)e * (HIDDEN / 16) + (hcol0 >> 4)) * (D_MODEL / 32)) * 256;
        __builtin_prefetch(W1f + 32 * 256, 0, 3);   // next W1 wave-chunk, near scope
        __builtin_prefetch(
            W2s + (((long)e * (D_MODEL / 16) + wave * 8) * (HIDDEN / 32) + hc * 4) * 256,
            0, 3);                                  // upcoming W2 fragments

        v8f hacc = (v8f){0.f,0.f,0.f,0.f,0.f,0.f,0.f,0.f};
        // 2-deep software pipeline on B (scheduler may re-sink; harmless)
        v16bf b0 = loadB(W1f, 0, lane);
        v16bf b1v = loadB(W1f, 1, lane);
#pragma unroll
        for (int kb = 0; kb < D_MODEL / 32; ++kb) {
            v16bf bn;
            if (kb + 2 < D_MODEL / 32) bn = loadB(W1f, kb + 2, lane);
            const v16bf a = packA(Xlds, D_MODEL, kb * 32, lane);
            hacc = __builtin_amdgcn_wmma_f32_16x16x32_bf16(
                false, a, false, b0, (short)0, hacc, false, false);
            b0 = b1v;
            b1v = bn;
        }
        // bias + ReLU, write 16x16 sub-tile into current H buffer
        {
            const int n  = lane & 15;
            const int mb = (lane & 16) ? 8 : 0;
            const float bv = b1[(size_t)e * HIDDEN + hcol0 + n];
#pragma unroll
            for (int r = 0; r < 8; ++r) {
                float hv = hacc[r] + bv;
                hv = hv > 0.0f ? hv : 0.0f;
                Hbuf[(mb + r) * 128 + wave * 16 + n] = (__bf16)hv;
            }
        }
        __syncthreads();   // single barrier per chunk (Hlds double-buffered)

        // ---- FFN2: out[:, wave*128 .. +128) += H_chunk @ W2[hc*128.., :], K = 128
#pragma unroll
        for (int sub = 0; sub < 8; ++sub) {
            const int nblk = wave * 8 + sub;   // out column block / 16
            const unsigned int* W2f =
                W2s + (((long)e * (D_MODEL / 16) + nblk) * (HIDDEN / 32) + hc * 4) * 256;
            // preload all 4 B fragments, then run the 4 WMMAs
            v16bf bb[4];
#pragma unroll
            for (int kk = 0; kk < 4; ++kk) bb[kk] = loadB(W2f, kk, lane);
            v8f c = cacc[sub];
#pragma unroll
            for (int kk = 0; kk < 4; ++kk) {
                const v16bf a = packA(Hbuf, 128, kk * 32, lane);
                c = __builtin_amdgcn_wmma_f32_16x16x32_bf16(
                    false, a, false, bb[kk], (short)0, c, false, false);
            }
            cacc[sub] = c;
        }
        // no trailing barrier: next chunk writes the other H buffer; the
        // barrier above guarantees all waves finished the previous buffer.
    }

    // ---- epilogue: + b2, scale by gate, scatter-add into output rows
    {
        const int n  = lane & 15;
        const int mb = (lane & 16) ? 8 : 0;
#pragma unroll
        for (int sub = 0; sub < 8; ++sub) {
            const int col  = wave * 128 + sub * 16 + n;
            const float bv = b2[(size_t)e * D_MODEL + col];
#pragma unroll
            for (int r = 0; r < 8; ++r) {
                const int m   = mb + r;
                const float g = grow[m];
                const float v = g * (cacc[sub][r] + bv);
                atomicAdd(&out[(size_t)trow[m] * D_MODEL + col], v);
            }
        }
    }
}

// ------------------------------------------------------------- launch ------
extern "C" void kernel_launch(void* const* d_in, const int* in_sizes, int n_in,
                              void* d_out, int out_size, void* d_ws, size_t ws_size,
                              hipStream_t stream) {
    const float* x  = (const float*)d_in[0];
    const float* Wg = (const float*)d_in[1];
    const float* bg = (const float*)d_in[2];
    const float* W1 = (const float*)d_in[3];
    const float* b1 = (const float*)d_in[4];
    const float* W2 = (const float*)d_in[5];
    const float* b2 = (const float*)d_in[6];
    float* out = (float*)d_out;

    // workspace layout
    char*  ws         = (char*)d_ws;
    int*   counts     = (int*)(ws + 0);      // 8 ints
    float* probs_sum  = (float*)(ws + 32);   // 8 floats
    float* assign_sum = (float*)(ws + 64);   // 8 floats
    int*   perm       = (int*)(ws + 256);                              // 8*8192 ints
    float* gatebuf    = (float*)(ws + 256 + NEXP * CAP * sizeof(int)); // 8*8192 floats

    const size_t W1S_OFF = (size_t)1 << 20;                          // 1 MB
    const size_t WB      = (size_t)NEXP * D_MODEL * HIDDEN * 2;      // 64 MB bf16
    unsigned int* W1s = (unsigned int*)(ws + W1S_OFF);
    unsigned int* W2s = (unsigned int*)(ws + W1S_OFF + WB);
    unsigned int* xs  = (unsigned int*)(ws + W1S_OFF + 2 * WB);      // 16 MB bf16

    const int out_n = NTOK * D_MODEL + 1 + 3 * NEXP;
    float* meta = out + (size_t)NTOK * D_MODEL;

    moe_zero_kernel<<<(out_n + 255) / 256, 256, 0, stream>>>(
        out, out_n, (unsigned int*)ws);

    // one-time fp32 -> bf16 conversions (weights fragment-swizzled; fits L2)
    const long totW = (long)NEXP * D_MODEL * HIDDEN / 2;    // dwords (2 bf16 each)
    moe_swz_kernel<<<32768, 256, 0, stream>>>(W1, W1s, D_MODEL, HIDDEN, totW);
    moe_swz_kernel<<<32768, 256, 0, stream>>>(W2, W2s, HIDDEN, D_MODEL, totW);
    moe_cvtx_kernel<<<8192, 256, 0, stream>>>(x, xs, (long)NTOK * D_MODEL / 2);

    moe_gate_kernel<<<NTOK / 256, 256, 0, stream>>>(
        x, Wg, bg, counts, probs_sum, assign_sum, perm, gatebuf);

    moe_finalize_kernel<<<1, 32, 0, stream>>>(counts, probs_sum, assign_sum, meta);

    dim3 grid(CAP / 16, NEXP);
    moe_ffn_kernel<<<grid, 256, 0, stream>>>(
        xs, perm, gatebuf, counts, W1s, b1, W2s, b2, out);
}
